// CrossAttention_75342316306879
// MI455X (gfx1250) — matrix operand order repaired and checked by reading, not compiled, not run on hardware
//
#include <hip/hip_runtime.h>

// Problem sizes (fixed by the reference).
#define BATCH 8
#define SEQ   2048
#define EMB   512

typedef __attribute__((ext_vector_type(16))) __bf16 v16bf;
typedef __attribute__((ext_vector_type(8)))  __bf16 v8bf;
typedef __attribute__((ext_vector_type(4)))  __bf16 v4bf;
typedef __attribute__((ext_vector_type(8)))  float  v8f;
typedef __attribute__((ext_vector_type(4)))  float  v4f;

// 128-bit payload type for the async-to-LDS builtin: clang declares it as a
// GCC-style vector (V4i) with addrspace(1) source / addrspace(3) destination.
typedef int v4i32 __attribute__((vector_size(16)));
typedef __attribute__((address_space(1))) v4i32 gv4i32;
typedef __attribute__((address_space(3))) v4i32 lv4i32;

#ifndef __has_builtin
#define __has_builtin(x) 0
#endif
#if __has_builtin(__builtin_amdgcn_global_load_async_to_lds_b128) && \
    __has_builtin(__builtin_amdgcn_s_wait_asynccnt)
#define HAVE_ASYNC_LDS 1
#else
#define HAVE_ASYNC_LDS 0
#endif

__device__ __forceinline__ __bf16 f2bf(float f) { return (__bf16)f; }

// ---------------------------------------------------------------------------
// CDNA5 WMMA fragment loaders (v_wmma_f32_16x16x32_bf16, wave32).
//
// A (16x32, MxK), row-major source with row stride `ld`:
//   lane m = lane&15; lanes 0-15 hold K 0..7 and 16..23, lanes 16-31 hold
//   K 8..15 and 24..31  -> two contiguous 16B loads per lane.
// B (32x16, KxN), source stored N-major (row n holds K contiguously, stride
//   `ld`): lane n = lane&15; lanes 0-15 hold K 0..7 and 16..23, lanes 16-31
//   hold K 8..15 and 24..31 -> two contiguous 16B loads per lane.
// C/D (16x16 f32): VGPR i holds row m = i + (lane<16 ? 0 : 8), col n = lane&15.
// ---------------------------------------------------------------------------
__device__ __forceinline__ v16bf load_a_frag(const __bf16* base, int ld, int lane) {
  const int m    = lane & 15;
  const int koff = (lane & 16) ? 8 : 0;
  const __bf16* p = base + (size_t)m * ld + koff;
  v8bf lo = *(const v8bf*)(p);
  v8bf hi = *(const v8bf*)(p + 16);
  v16bf r;
#pragma unroll
  for (int i = 0; i < 8; ++i) { r[i] = lo[i]; r[i + 8] = hi[i]; }
  return r;
}

__device__ __forceinline__ v16bf load_b_frag(const __bf16* base, int ld, int lane) {
  const int n    = lane & 15;
  const int koff = (lane & 16) ? 8 : 0;
  const __bf16* p = base + (size_t)n * ld + koff;
  v8bf lo = *(const v8bf*)(p);
  v8bf hi = *(const v8bf*)(p + 16);
  v16bf r;
#pragma unroll
  for (int i = 0; i < 8; ++i) { r[i] = lo[i]; r[i + 8] = hi[i]; }
  return r;
}

__device__ __forceinline__ v8f wmma_bf16(v16bf a, v16bf b, v8f c) {
  return __builtin_amdgcn_wmma_f32_16x16x32_bf16(false, a, false, b,
                                                 (short)0, c, false, false);
}

// ---------------------------------------------------------------------------
// K0: f32 -> bf16 cast (4 elems / thread)
// ---------------------------------------------------------------------------
__global__ void cast_bf16_kernel(const float* __restrict__ in,
                                 __bf16* __restrict__ out, int n4) {
  int i = blockIdx.x * blockDim.x + threadIdx.x;
  if (i >= n4) return;
  v4f v = *(const v4f*)(in + 4 * (size_t)i);
  v4bf o;
#pragma unroll
  for (int j = 0; j < 4; ++j) o[j] = f2bf(v[j]);
  *(v4bf*)(out + 4 * (size_t)i) = o;
}

// ---------------------------------------------------------------------------
// K1: projection GEMM  out[m,f] = sum_e X[m,e] * W[f,e]
// M = BATCH*SEQ, N = K = EMB. Each wave computes a 16x64 strip (4 tiles):
// one A fragment feeds 4 WMMAs / 4 accumulator chains. Software-pipelined:
// fragments for k-step ks+1 are issued before the WMMAs of step ks so VMEM
// overlaps the XDL pipe (partial s_wait_loadcnt instead of full drains).
// ---------------------------------------------------------------------------
__global__ void __launch_bounds__(256)
proj_gemm_kernel(const __bf16* __restrict__ X, const __bf16* __restrict__ W,
                 __bf16* __restrict__ out_bf, float* __restrict__ out_f32,
                 __bf16* __restrict__ out_T) {
  const int lane = threadIdx.x & 31;
  const int wid  = threadIdx.x >> 5;
  const int wave = blockIdx.x * 8 + wid;
  const int mt   = wave >> 3;      // EMB/64 = 8 n-groups per m-tile row
  const int ng   = wave & 7;
  const int m0   = mt * 16;
  const int n0   = ng * 64;

  const __bf16* abase = X + (size_t)m0 * EMB;
  const __bf16* bbase = W + (size_t)n0 * EMB;

  // prologue: fragments for ks = 0
  v16bf a = load_a_frag(abase, EMB, lane);
  v16bf b[4];
#pragma unroll
  for (int j = 0; j < 4; ++j)
    b[j] = load_b_frag(bbase + (size_t)j * 16 * EMB, EMB, lane);

  v8f acc[4] = {{}, {}, {}, {}};
#pragma unroll 2
  for (int ks = 0; ks < EMB / 32 - 1; ++ks) {
    v16bf an = load_a_frag(abase + (ks + 1) * 32, EMB, lane);
    v16bf bn[4];
#pragma unroll
    for (int j = 0; j < 4; ++j)
      bn[j] = load_b_frag(bbase + (size_t)j * 16 * EMB + (ks + 1) * 32, EMB, lane);
#pragma unroll
    for (int j = 0; j < 4; ++j) acc[j] = wmma_bf16(a, b[j], acc[j]);
    a = an;
#pragma unroll
    for (int j = 0; j < 4; ++j) b[j] = bn[j];
  }
#pragma unroll
  for (int j = 0; j < 4; ++j) acc[j] = wmma_bf16(a, b[j], acc[j]);  // epilogue

  const int moff = (lane & 16) ? 8 : 0;
  const int n    = lane & 15;
#pragma unroll
  for (int j = 0; j < 4; ++j) {
    const int nf = n0 + j * 16 + n;
    if (out_bf) {
#pragma unroll
      for (int i = 0; i < 8; ++i)
        out_bf[(size_t)(m0 + moff + i) * EMB + nf] = f2bf(acc[j][i]);
    }
    if (out_f32) {
#pragma unroll
      for (int i = 0; i < 8; ++i)
        out_f32[(size_t)(m0 + moff + i) * EMB + nf] = acc[j][i];
    }
    if (out_T) {  // vT[b][f][s]: one contiguous 16B store per lane
      const int bb = m0 >> 11;            // m0 / SEQ
      const int s0 = m0 & (SEQ - 1);
      v8bf pk;
#pragma unroll
      for (int i = 0; i < 8; ++i) pk[i] = f2bf(acc[j][i]);
      *(v8bf*)(out_T + ((size_t)bb * EMB + nf) * SEQ + s0 + moff) = pk;
    }
  }
}

// ---------------------------------------------------------------------------
// K2: fused attention. One workgroup (8 waves) per (batch, 16-row q block).
// Entire 16x2048 score strip lives in LDS (CDNA5: 320KB/WGP).
//   phase 1: scores = q_tile @ k^T  (q A-fragments hoisted into 128 VGPRs,
//            k B-fragments software-pipelined)
//   phase 2: scale + pad mask + softmax (unnormalized probs -> bf16, keep 1/l)
//   phase 3: O = P @ vT              (A from LDS + B from global, pipelined)
//   phase 4: OUT = diag(1/l) * (O @ Wp^T) + bp + q   (residual read from d_out)
// ---------------------------------------------------------------------------
__global__ void __launch_bounds__(256)
attn_kernel(const __bf16* __restrict__ qbf, const __bf16* __restrict__ kbf,
            const __bf16* __restrict__ vT, const __bf16* __restrict__ Wpb,
            const float* __restrict__ bp, const unsigned char* __restrict__ pad,
            float* __restrict__ out) {
  extern __shared__ char smem[];
  float*  sc   = (float*)smem;                      // [16][2048] f32   128KB
  __bf16* pr   = (__bf16*)(smem + 131072);          // [16][2048] bf16   64KB
  __bf16* qt   = (__bf16*)(smem + 196608);          // [16][512]  bf16   16KB
  __bf16* ob   = (__bf16*)(smem + 212992);          // [16][512]  bf16   16KB
  float*  redm = (float*)(smem + 229376);           // [8]
  float*  reds = redm + 8;                          // [8]
  float*  linv = reds + 8;                          // [16]

  const int tid  = threadIdx.x;
  const int lane = tid & 31;
  const int wid  = tid >> 5;
  const int b    = blockIdx.x >> 7;                 // SEQ/16 = 128 q-blocks
  const int q0   = (blockIdx.x & 127) * 16;

  // ---- stage q tile (16x512 bf16 = 16KB) into LDS via async DMA ----
  {
    const __bf16* src = qbf + ((size_t)b * SEQ + q0) * EMB;
#if HAVE_ASYNC_LDS
#pragma unroll
    for (int i = 0; i < 4; ++i) {
      int idx = (tid + i * 256) * 8;                // 16B chunks
      __builtin_amdgcn_global_load_async_to_lds_b128(
          (gv4i32*)(src + idx), (lv4i32*)(qt + idx), 0, 0);
    }
    __builtin_amdgcn_s_wait_asynccnt(0);
#else
#pragma unroll
    for (int i = 0; i < 4; ++i) {
      int idx = (tid + i * 256) * 8;
      *(v8bf*)(qt + idx) = *(const v8bf*)(src + idx);
    }
#endif
  }
  __syncthreads();

  const float scale = 0.044194173824159216f;        // EMB^-0.5

  // ---- phase 1: scores = q @ k^T ----
  // Hoist all 16 q A-fragments into registers (16 x 8 VGPRs = 128 VGPRs).
  v16bf aq[16];
#pragma unroll
  for (int ks = 0; ks < EMB / 32; ++ks)
    aq[ks] = load_a_frag(qt + ks * 32, EMB, lane);

  const __bf16* kb = kbf + (size_t)b * SEQ * EMB;
#pragma unroll 1
  for (int t = 0; t < 16; ++t) {
    const int nt = wid + t * 8;                     // 0..127
    const __bf16* bbase = kb + (size_t)nt * 16 * EMB;
    __builtin_prefetch((const void*)(kb + (size_t)(nt + 8) * 16 * EMB +
                                     (size_t)(lane & 15) * EMB), 0, 3);
    v8f acc = {};
    v16bf bb = load_b_frag(bbase, EMB, lane);
#pragma unroll
    for (int ks = 0; ks < EMB / 32 - 1; ++ks) {
      v16bf nb = load_b_frag(bbase + (ks + 1) * 32, EMB, lane);
      acc = wmma_bf16(aq[ks], bb, acc);
      bb = nb;
    }
    acc = wmma_bf16(aq[EMB / 32 - 1], bb, acc);     // epilogue
    const int moff = (lane & 16) ? 8 : 0;
    const int n    = lane & 15;
#pragma unroll
    for (int i = 0; i < 8; ++i)
      sc[(size_t)(moff + i) * SEQ + nt * 16 + n] = acc[i];
  }
  __syncthreads();

  // ---- phase 2: masked scaled softmax (per row; 256 threads x 8 cols) ----
  for (int r = 0; r < 16; ++r) {
    float* row = sc + (size_t)r * SEQ;
    const unsigned char* mrow = pad + ((size_t)b * SEQ + q0 + r) * SEQ;
    float lmax = -3.0e38f;
#pragma unroll
    for (int i = 0; i < 8; ++i) {
      int c = tid + i * 256;
      float v = row[c] * scale;
      if (mrow[c]) v = -1.0e9f;                     // jnp.where(pad_mask, -1e9, s)
      row[c] = v;
      lmax = fmaxf(lmax, v);
    }
#pragma unroll
    for (int off = 16; off > 0; off >>= 1)
      lmax = fmaxf(lmax, __shfl_xor(lmax, off, 32));
    if (lane == 0) redm[wid] = lmax;
    __syncthreads();
    float rmax = redm[0];
#pragma unroll
    for (int w = 1; w < 8; ++w) rmax = fmaxf(rmax, redm[w]);
    float lsum = 0.f;
    __bf16* prow = pr + (size_t)r * SEQ;
#pragma unroll
    for (int i = 0; i < 8; ++i) {
      int c = tid + i * 256;
      float e = __expf(row[c] - rmax);
      prow[c] = f2bf(e);                            // unnormalized prob
      lsum += e;
    }
#pragma unroll
    for (int off = 16; off > 0; off >>= 1)
      lsum += __shfl_xor(lsum, off, 32);
    if (lane == 0) reds[wid] = lsum;
    __syncthreads();
    if (tid == 0) {
      float s = 0.f;
#pragma unroll
      for (int w = 0; w < 8; ++w) s += reds[w];
      linv[r] = 1.0f / s;
    }
  }
  __syncthreads();

  // ---- phase 3: O = P @ V (uses vT so B-fragments are contiguous) ----
  const __bf16* vb = vT + (size_t)b * EMB * SEQ;
#pragma unroll 1
  for (int t = 0; t < 4; ++t) {
    const int nt = wid + t * 8;                     // 0..31 over EMB
    const __bf16* bbase = vb + (size_t)nt * 16 * SEQ;
    v8f acc = {};
    v16bf a  = load_a_frag(pr, SEQ, lane);
    v16bf bb = load_b_frag(bbase, SEQ, lane);
#pragma unroll 4
    for (int js = 0; js < SEQ / 32 - 1; ++js) {
      v16bf na = load_a_frag(pr + (js + 1) * 32, SEQ, lane);
      v16bf nb = load_b_frag(bbase + (js + 1) * 32, SEQ, lane);
      acc = wmma_bf16(a, bb, acc);
      a = na; bb = nb;
    }
    acc = wmma_bf16(a, bb, acc);                    // epilogue
    const int moff = (lane & 16) ? 8 : 0;
    const int n    = lane & 15;
#pragma unroll
    for (int i = 0; i < 8; ++i)
      ob[(size_t)(moff + i) * EMB + nt * 16 + n] = f2bf(acc[i]);
  }
  __syncthreads();

  // ---- phase 4: output projection + bias + residual ----
#pragma unroll 1
  for (int t = 0; t < 4; ++t) {
    const int nt = wid + t * 8;
    const __bf16* bbase = Wpb + (size_t)nt * 16 * EMB;
    v8f acc = {};
    v16bf a  = load_a_frag(ob, EMB, lane);
    v16bf bb = load_b_frag(bbase, EMB, lane);
#pragma unroll
    for (int ks = 0; ks < EMB / 32 - 1; ++ks) {
      v16bf na = load_a_frag(ob + (ks + 1) * 32, EMB, lane);
      v16bf nb = load_b_frag(bbase + (ks + 1) * 32, EMB, lane);
      acc = wmma_bf16(a, bb, acc);
      a = na; bb = nb;
    }
    acc = wmma_bf16(a, bb, acc);                    // epilogue
    const int moff = (lane & 16) ? 8 : 0;
    const int n    = lane & 15;
    const int nf   = nt * 16 + n;
    const float bias = bp[nf];
#pragma unroll
    for (int i = 0; i < 8; ++i) {
      int m = moff + i;
      size_t idx = ((size_t)b * SEQ + q0 + m) * EMB + nf;
      out[idx] = acc[i] * linv[m] + bias + out[idx];  // out held q (f32) from K1
    }
  }
}

// ---------------------------------------------------------------------------
extern "C" void kernel_launch(void* const* d_in, const int* in_sizes, int n_in,
                              void* d_out, int out_size, void* d_ws, size_t ws_size,
                              hipStream_t stream) {
  (void)in_sizes; (void)n_in; (void)out_size; (void)ws_size;
  const float* Q  = (const float*)d_in[0];
  const float* K  = (const float*)d_in[1];
  // jnp bool arrays are 1 byte per element
  const unsigned char* pad = (const unsigned char*)d_in[2];
  const float* Wq = (const float*)d_in[3];
  const float* Wk = (const float*)d_in[4];
  const float* Wv = (const float*)d_in[5];
  const float* Wp = (const float*)d_in[6];
  const float* bp = (const float*)d_in[7];
  float* out = (float*)d_out;

  const size_t NSE = (size_t)BATCH * SEQ * EMB;     // 8,388,608
  __bf16* Qbf = (__bf16*)d_ws;
  __bf16* Kbf = Qbf + NSE;
  __bf16* qbf = Kbf + NSE;
  __bf16* kbf = qbf + NSE;
  __bf16* vT  = kbf + NSE;
  __bf16* Wqb = vT + NSE;
  __bf16* Wkb = Wqb + (size_t)EMB * EMB;
  __bf16* Wvb = Wkb + (size_t)EMB * EMB;
  __bf16* Wpb = Wvb + (size_t)EMB * EMB;

  // casts
  const int n4a = (int)(NSE / 4);                   // activations
  const int n4w = EMB * EMB / 4;                    // weights
  cast_bf16_kernel<<<n4a / 256, 256, 0, stream>>>(Q, Qbf, n4a);
  cast_bf16_kernel<<<n4a / 256, 256, 0, stream>>>(K, Kbf, n4a);
  cast_bf16_kernel<<<n4w / 256, 256, 0, stream>>>(Wq, Wqb, n4w);
  cast_bf16_kernel<<<n4w / 256, 256, 0, stream>>>(Wk, Wkb, n4w);
  cast_bf16_kernel<<<n4w / 256, 256, 0, stream>>>(Wv, Wvb, n4w);
  cast_bf16_kernel<<<n4w / 256, 256, 0, stream>>>(Wp, Wpb, n4w);

  // projections: waves = (M/16)*(N/64) = 1024*8; 8 waves/block -> 1024 blocks
  proj_gemm_kernel<<<1024, 256, 0, stream>>>(Qbf, Wqb, qbf, out, nullptr);
  proj_gemm_kernel<<<1024, 256, 0, stream>>>(Kbf, Wkb, kbf, nullptr, nullptr);
  proj_gemm_kernel<<<1024, 256, 0, stream>>>(Kbf, Wvb, nullptr, nullptr, vT);

  // fused attention: one block per (batch, 16-row q block)
  const int shmem = 229504;                         // ~224KB of CDNA5's 320KB LDS
  (void)hipFuncSetAttribute((const void*)attn_kernel,
                            hipFuncAttributeMaxDynamicSharedMemorySize, shmem);
  attn_kernel<<<BATCH * (SEQ / 16), 256, shmem, stream>>>(qbf, kbf, vT, Wpb,
                                                          bp, pad, out);
}
